// AttentionLayer_41386304864292
// MI455X (gfx1250) — matrix operand order repaired and checked by reading, not compiled, not run on hardware
//
#include <hip/hip_runtime.h>
#include <hip/hip_bf16.h>

typedef __bf16 bf16;
typedef __attribute__((ext_vector_type(16))) __bf16 v16bf;
typedef __attribute__((ext_vector_type(8)))  __bf16 v8bf;
typedef __attribute__((ext_vector_type(8)))  float  v8f;

#define BATCH  8
#define CCH    256
#define NPIX   16384
#define KSPLIT 8
#define LSTR   40   // LDS row stride in bf16 elements (80B, conflict-free, 16B aligned)

__device__ inline v16bf pack16(v8bf lo, v8bf hi) {
  v16bf r;
#pragma unroll
  for (int e = 0; e < 8; ++e) { r[e] = lo[e]; r[e + 8] = hi[e]; }
  return r;
}

__device__ inline v8bf cvt8(float4 a, float4 b) {
  v8bf r;
  r[0] = (bf16)a.x; r[1] = (bf16)a.y; r[2] = (bf16)a.z; r[3] = (bf16)a.w;
  r[4] = (bf16)b.x; r[5] = (bf16)b.y; r[6] = (bf16)b.z; r[7] = (bf16)b.w;
  return r;
}

// ---------------------------------------------------------------------------
// Kernel 1: E_part[ks][b] += q_tile . q_tile^T   (64x64 output tile, K-chunk 2048)
// grid = (16 tiles, 8 batch, 8 ksplit), block = 256 (8 waves)
// ---------------------------------------------------------------------------
__global__ __launch_bounds__(256) void energy_kernel(const float* __restrict__ x,
                                                     float* __restrict__ epart) {
  __shared__ bf16 ldsA[64 * LSTR];
  __shared__ bf16 ldsB[64 * LSTR];

  const int tile = blockIdx.x;
  const int bb   = blockIdx.y;
  const int ks   = blockIdx.z;
  const int i0 = (tile >> 2) * 64;
  const int j0 = (tile & 3) * 64;

  const int t    = threadIdx.x;
  const int lane = t & 31;
  const int w    = t >> 5;
  const int m    = w & 3;        // M subtile of this wave
  const int np   = w >> 2;       // N subtile pair (0/1)
  const int lrow = lane & 15;
  const int hs   = lane >> 4;

  const float* qb = x + (size_t)bb * CCH * NPIX;

  v8f acc0 = {};
  v8f acc1 = {};

  const int rowL = t >> 2;        // 0..63 staging row
  const int cg   = (t & 3) * 8;   // 0,8,16,24 staging col group

  for (int it = 0; it < 2048 / 32; ++it) {
    const int kk = ks * 2048 + it * 32;

    const float* pa = qb + (size_t)(i0 + rowL) * NPIX + kk + cg;
    const float* pb = qb + (size_t)(j0 + rowL) * NPIX + kk + cg;
    float4 a0 = *(const float4*)(pa);
    float4 a1 = *(const float4*)(pa + 4);
    float4 b0 = *(const float4*)(pb);
    float4 b1 = *(const float4*)(pb + 4);

    __syncthreads();
    *(v8bf*)&ldsA[rowL * LSTR + cg] = cvt8(a0, a1);
    *(v8bf*)&ldsB[rowL * LSTR + cg] = cvt8(b0, b1);
    __syncthreads();

    // A fragment: contiguous K half per lane (matches B)
    const int ar = m * 16 + lrow;
    v8bf alo = *(const v8bf*)&ldsA[ar * LSTR + hs * 16];
    v8bf ahi = *(const v8bf*)&ldsA[ar * LSTR + hs * 16 + 8];
    v16bf af = pack16(alo, ahi);

    // B fragments: contiguous K half per lane (B layout)
    const int br0 = (np * 2 + 0) * 16 + lrow;
    const int br1 = (np * 2 + 1) * 16 + lrow;
    v8bf b0lo = *(const v8bf*)&ldsB[br0 * LSTR + hs * 16];
    v8bf b0hi = *(const v8bf*)&ldsB[br0 * LSTR + hs * 16 + 8];
    v8bf b1lo = *(const v8bf*)&ldsB[br1 * LSTR + hs * 16];
    v8bf b1hi = *(const v8bf*)&ldsB[br1 * LSTR + hs * 16 + 8];
    v16bf bf0 = pack16(b0lo, b0hi);
    v16bf bf1 = pack16(b1lo, b1hi);

    acc0 = __builtin_amdgcn_wmma_f32_16x16x32_bf16(false, af, false, bf0,
                                                   (short)0, acc0, false, false);
    acc1 = __builtin_amdgcn_wmma_f32_16x16x32_bf16(false, af, false, bf1,
                                                   (short)0, acc1, false, false);
  }

  float* dst = epart + (size_t)(ks * BATCH + bb) * CCH * CCH;
#pragma unroll
  for (int r = 0; r < 8; ++r) {
    const int row = i0 + m * 16 + r + 8 * hs;
    const int c0  = j0 + (np * 2 + 0) * 16 + lrow;
    dst[row * CCH + c0]      = acc0[r];
    dst[row * CCH + c0 + 16] = acc1[r];
  }
}

// ---------------------------------------------------------------------------
// Kernel 2: sum K-split partials + row softmax.  One block per (b, i) row.
// ---------------------------------------------------------------------------
__global__ __launch_bounds__(256) void softmax_kernel(const float* __restrict__ epart,
                                                      float* __restrict__ attn) {
  __shared__ float red[256];
  const int b = blockIdx.x >> 8;
  const int i = blockIdx.x & 255;
  const int j = threadIdx.x;

  float v = 0.f;
#pragma unroll
  for (int ks = 0; ks < KSPLIT; ++ks)
    v += epart[(size_t)(ks * BATCH + b) * CCH * CCH + i * CCH + j];

  red[j] = v;
  __syncthreads();
  for (int s = 128; s > 0; s >>= 1) {
    if (j < s) red[j] = fmaxf(red[j], red[j + s]);
    __syncthreads();
  }
  const float mx = red[0];
  __syncthreads();

  const float e = __expf(v - mx);
  red[j] = e;
  __syncthreads();
  for (int s = 128; s > 0; s >>= 1) {
    if (j < s) red[j] += red[j + s];
    __syncthreads();
  }
  attn[(size_t)b * CCH * CCH + i * CCH + j] = e * (1.0f / red[0]);
}

// ---------------------------------------------------------------------------
// Kernel 3: M = w2 @ attn  (tiny: ~268 MFLOP total), bf16 output.
// One block per (b, o); thread j sweeps i with coalesced attn reads.
// ---------------------------------------------------------------------------
__global__ __launch_bounds__(256) void mmat_kernel(const float* __restrict__ w2,
                                                   const float* __restrict__ attn,
                                                   bf16* __restrict__ mmat) {
  const int b = blockIdx.x >> 8;
  const int o = blockIdx.x & 255;
  const int j = threadIdx.x;
  const float* arow = attn + (size_t)b * CCH * CCH;
  const float* wrow = w2 + o * CCH;
  float acc = 0.f;
#pragma unroll 4
  for (int i = 0; i < CCH; ++i) acc += wrow[i] * arow[i * CCH + j];
  mmat[(size_t)b * CCH * CCH + o * CCH + j] = (bf16)acc;
}

// ---------------------------------------------------------------------------
// Kernel 4: out = M @ q + bias.  WG computes 256(o) x 64(n), K=256.
// grid = (256 n-tiles, 8 batch), block = 256 (8 waves; wave w owns 32 o-rows)
// ---------------------------------------------------------------------------
__global__ __launch_bounds__(256) void outgemm_kernel(const float* __restrict__ x,
                                                      const bf16* __restrict__ mmat,
                                                      const float* __restrict__ bias,
                                                      float* __restrict__ out) {
  __shared__ bf16 ldsB[64 * LSTR];   // [n-col][K] transposed q tile

  const int b  = blockIdx.y;
  const int n0 = blockIdx.x * 64;
  const int t    = threadIdx.x;
  const int lane = t & 31;
  const int w    = t >> 5;
  const int lrow = lane & 15;
  const int hs   = lane >> 4;

  const float* qb = x + (size_t)b * CCH * NPIX;
  const bf16*  mb = mmat + (size_t)b * CCH * CCH;

  v8f acc[2][4] = {};

  const int jr = t >> 3;        // 0..31 staging K-row
  const int cg = (t & 7) * 8;   // 0..56 staging n-col group

  for (int k0 = 0; k0 < CCH; k0 += 32) {
    const float* pq = qb + (size_t)(k0 + jr) * NPIX + n0 + cg;
    float4 q0 = *(const float4*)(pq);
    float4 q1 = *(const float4*)(pq + 4);

    __syncthreads();
    // transpose-on-store: LDS is [col][K]
    ldsB[(cg + 0) * LSTR + jr] = (bf16)q0.x;
    ldsB[(cg + 1) * LSTR + jr] = (bf16)q0.y;
    ldsB[(cg + 2) * LSTR + jr] = (bf16)q0.z;
    ldsB[(cg + 3) * LSTR + jr] = (bf16)q0.w;
    ldsB[(cg + 4) * LSTR + jr] = (bf16)q1.x;
    ldsB[(cg + 5) * LSTR + jr] = (bf16)q1.y;
    ldsB[(cg + 6) * LSTR + jr] = (bf16)q1.z;
    ldsB[(cg + 7) * LSTR + jr] = (bf16)q1.w;
    __syncthreads();

    v16bf bfr[4];
#pragma unroll
    for (int nn = 0; nn < 4; ++nn) {
      const int col = nn * 16 + lrow;
      v8bf lo = *(const v8bf*)&ldsB[col * LSTR + hs * 16];
      v8bf hi = *(const v8bf*)&ldsB[col * LSTR + hs * 16 + 8];
      bfr[nn] = pack16(lo, hi);
    }

#pragma unroll
    for (int mm = 0; mm < 2; ++mm) {
      const int orow = w * 32 + mm * 16 + lrow;
      const bf16* ap = mb + orow * CCH + k0 + hs * 16;   // contiguous A half
      v8bf alo = *(const v8bf*)(ap);
      v8bf ahi = *(const v8bf*)(ap + 8);
      v16bf af = pack16(alo, ahi);
#pragma unroll
      for (int nn = 0; nn < 4; ++nn)
        acc[mm][nn] = __builtin_amdgcn_wmma_f32_16x16x32_bf16(
            false, af, false, bfr[nn], (short)0, acc[mm][nn], false, false);
    }
  }

  float* ob = out + (size_t)b * CCH * NPIX;
#pragma unroll
  for (int mm = 0; mm < 2; ++mm) {
#pragma unroll
    for (int nn = 0; nn < 4; ++nn) {
#pragma unroll
      for (int r = 0; r < 8; ++r) {
        const int o = w * 32 + mm * 16 + r + 8 * hs;
        const int n = n0 + nn * 16 + lrow;
        ob[(size_t)o * NPIX + n] = acc[mm][nn][r] + bias[o];
      }
    }
  }
}

// ---------------------------------------------------------------------------
extern "C" void kernel_launch(void* const* d_in, const int* in_sizes, int n_in,
                              void* d_out, int out_size, void* d_ws, size_t ws_size,
                              hipStream_t stream) {
  const float* x      = (const float*)d_in[0];
  const float* conv_w = (const float*)d_in[1];   // (256,256,1,1)
  const float* conv_b = (const float*)d_in[2];   // (256,)
  float* out = (float*)d_out;

  char* ws = (char*)d_ws;
  float* epart = (float*)ws;                                       // 8*8*256*256*4 = 16 MB
  float* attn  = (float*)(ws + (size_t)KSPLIT * BATCH * CCH * CCH * 4);          // 2 MB
  bf16*  mmat  = (bf16*)(ws + (size_t)KSPLIT * BATCH * CCH * CCH * 4
                            + (size_t)BATCH * CCH * CCH * 4);                    // 1 MB

  energy_kernel <<<dim3(16, BATCH, KSPLIT), 256, 0, stream>>>(x, epart);
  softmax_kernel<<<BATCH * CCH,            256, 0, stream>>>(epart, attn);
  mmat_kernel   <<<BATCH * CCH,            256, 0, stream>>>(conv_w, attn, mmat);
  outgemm_kernel<<<dim3(NPIX / 64, BATCH), 256, 0, stream>>>(x, mmat, conv_b, out);
}
